// GraphAttentionLayer_63333587747451
// MI455X (gfx1250) — matrix operand order repaired and checked by reading, not compiled, not run on hardware
//
#include <hip/hip_runtime.h>
#include <hip/hip_bf16.h>

typedef __attribute__((ext_vector_type(16))) _Float16 v16h;
typedef __attribute__((ext_vector_type(8)))  _Float16 v8h;
typedef __attribute__((ext_vector_type(8)))  float    v8f;
typedef __attribute__((ext_vector_type(4)))  unsigned int u32x4;
typedef __attribute__((ext_vector_type(8)))  int          i32x8;
typedef __attribute__((ext_vector_type(4)))  int          i32x4;

#define ALPHA_ 0.2f

constexpr int B_ = 8, N_ = 1024, FIN = 64, T_ = 8, O_ = 128, BT_ = 64;

// Stage-3 LDS staging geometry: TDM copies a 64(j) x 128(o) f16 tile per chunk,
// padding +16B after every 128B row -> LDS row stride 72 f16 (bank-spread).
constexpr int JCHUNK = 64;          // j's per staged chunk
constexpr int LROW   = JCHUNK + 8;  // padded LDS row stride in f16 (144B)

// ---------------------------------------------------------------------------
// Stage 1: Wh[bt][n][o] = sum_f x[b][n][f][t] * W[f][o]   (f16 WMMA, f32 acc)
// Writes WhT (f16, [bt][o][n]) and projections Wh1 = Wh@a1, Wh2 = Wh@a2.
// ---------------------------------------------------------------------------
__global__ __launch_bounds__(256) void gat_wh_kernel(
    const float* __restrict__ x, const float* __restrict__ W,
    const float* __restrict__ a,
    _Float16* __restrict__ WhT, float* __restrict__ Wh1, float* __restrict__ Wh2)
{
  __shared__ _Float16 Wlds[FIN * O_];          // transposed: Wlds[o*64 + f]
  for (int idx = threadIdx.x; idx < FIN * O_; idx += 256) {
    int o = idx >> 6, f = idx & 63;
    Wlds[idx] = (_Float16)W[f * O_ + o];
  }
  __syncthreads();

  const int wid  = threadIdx.x >> 5;
  const int lane = threadIdx.x & 31;
  const int hf   = lane >> 4;
  const int l16  = lane & 15;
  const int gw   = blockIdx.x * 8 + wid;       // 0..4095
  const int b    = gw >> 9;
  const int rem  = gw & 511;
  const int t    = rem >> 6;
  const int ntile = rem & 63;
  const int bt   = b * T_ + t;
  const int nn   = ntile * 16 + l16;           // A row M

  const float* xrow = x + ((size_t)(b * N_ + nn) * FIN) * T_ + t;

  v8f acc[8] = {};

  #pragma unroll
  for (int fb = 0; fb < FIN; fb += 32) {
    v16h A;
    #pragma unroll
    for (int v = 0; v < 8; ++v) {
      int k0 = 2 * v + 16 * hf;
      A[2 * v]     = (_Float16)xrow[(size_t)(fb + k0) * T_];
      A[2 * v + 1] = (_Float16)xrow[(size_t)(fb + k0 + 1) * T_];
    }
    #pragma unroll
    for (int ot = 0; ot < 8; ++ot) {
      int o = ot * 16 + l16;
      v16h Bv;
      #pragma unroll
      for (int v = 0; v < 8; ++v) {
        int f = fb + 2 * v + 16 * hf;
        Bv[2 * v]     = Wlds[o * FIN + f];
        Bv[2 * v + 1] = Wlds[o * FIN + f + 1];
      }
      acc[ot] = __builtin_amdgcn_wmma_f32_16x16x32_f16(
          false, A, false, Bv, (short)0, acc[ot], false, false);
    }
  }

  float wh1p[8], wh2p[8];
  #pragma unroll
  for (int v = 0; v < 8; ++v) { wh1p[v] = 0.f; wh2p[v] = 0.f; }

  const int nbase = ntile * 16;
  #pragma unroll
  for (int ot = 0; ot < 8; ++ot) {
    int o = ot * 16 + l16;
    float a1v = a[o], a2v = a[O_ + o];
    v8h hv;
    #pragma unroll
    for (int v = 0; v < 8; ++v) {
      float c = acc[ot][v];
      hv[v] = (_Float16)c;
      wh1p[v] += c * a1v;
      wh2p[v] += c * a2v;
    }
    _Float16* dst = WhT + ((size_t)bt * O_ + o) * N_ + nbase + 8 * hf;
    *(v8h*)dst = hv;
  }
  #pragma unroll
  for (int mask = 1; mask <= 8; mask <<= 1) {
    #pragma unroll
    for (int v = 0; v < 8; ++v) {
      wh1p[v] += __shfl_xor(wh1p[v], mask, 32);
      wh2p[v] += __shfl_xor(wh2p[v], mask, 32);
    }
  }
  if (l16 == 0) {
    #pragma unroll
    for (int v = 0; v < 8; ++v) {
      int n = nbase + 8 * hf + v;
      Wh1[(size_t)bt * N_ + n] = wh1p[v];
      Wh2[(size_t)bt * N_ + n] = wh2p[v];
    }
  }
}

// ---------------------------------------------------------------------------
// Stage 2: per-row softmax stats over e_ij = mask(lrelu(Wh1[i]+Wh2[j])).
// ---------------------------------------------------------------------------
__global__ __launch_bounds__(256) void gat_stats_kernel(
    const float* __restrict__ adj, const float* __restrict__ Wh1,
    const float* __restrict__ Wh2, float* __restrict__ Mrow,
    float* __restrict__ Rrow)
{
  const int wid = threadIdx.x >> 5, lane = threadIdx.x & 31;
  const int w  = blockIdx.x * 8 + wid;         // 0..65535
  const int bt = w >> 10, i = w & (N_ - 1);
  const float* adjr = adj + (size_t)i * N_;
  const float* wh2r = Wh2 + (size_t)bt * N_;
  const float wh1i  = Wh1[(size_t)bt * N_ + i];

  float m = -3.0e38f;
  for (int j = lane; j < N_; j += 32) {
    float e = wh1i + wh2r[j];
    e = fmaxf(e, ALPHA_ * e);
    if (adjr[j] > 0.f) m = fmaxf(m, e);
  }
  #pragma unroll
  for (int mask = 16; mask >= 1; mask >>= 1) m = fmaxf(m, __shfl_xor(m, mask, 32));

  float s = 0.f;
  for (int j = lane; j < N_; j += 32) {
    float e = wh1i + wh2r[j];
    e = fmaxf(e, ALPHA_ * e);
    s += (adjr[j] > 0.f) ? __expf(e - m) : 0.f;
  }
  #pragma unroll
  for (int mask = 16; mask >= 1; mask >>= 1) s += __shfl_xor(s, mask, 32);

  if (lane == 0) {
    Mrow[(size_t)bt * N_ + i] = m;
    Rrow[(size_t)bt * N_ + i] = 1.f / s;
  }
}

// ---------------------------------------------------------------------------
// TDM: DMA one 64(j) x 128(o) f16 tile of WhT[bt] into LDS.
// D# group0: count=1 | lds_addr | global_addr | type=2.
// D# group1: data_size=2B, pad_enable, pad_interval=4 (pad each 128B row),
//            pad_amount=3 (+16B), tensor_dims 1024x128, tile 64x128,
//            dim0_stride=1024.  Groups 2/3 zero (<=2-D tensor).
// This toolchain exposes the 6-arg builtin (extra int32x8 before cpol).
// ---------------------------------------------------------------------------
__device__ inline void tdm_load_chunk(const _Float16* gsrc, unsigned int lds_byte_off)
{
  unsigned long long ga = (unsigned long long)(uintptr_t)gsrc;
  u32x4 g0 = { 1u,                                   // count=1, user mode
               lds_byte_off,                         // LDS dest (bytes)
               (unsigned int)ga,
               (unsigned int)(ga >> 32) | 0x80000000u };  // type=2 in [127:126]
  i32x8 g1 = { (int)(0x00010000u | (1u << 20) | (4u << 22) | (3u << 25)),
               (int)0x04000000u,    // tensor_dim0 = 1024 (bits 63:48)
               (int)0x00800000u,    // tensor_dim1 = 128  (bits 95:80)
               (int)0x00400000u,    // tile_dim0   = 64   (bits 127:112)
               128,                 // tile_dim1   = 128  (bits 143:128)
               1024,                // tensor_dim0_stride (bits 191:160)
               0, 0 };
  i32x4 z4 = { 0, 0, 0, 0 };
  i32x8 z8 = { 0, 0, 0, 0, 0, 0, 0, 0 };
  __builtin_amdgcn_tensor_load_to_lds(g0, g1, z4, z4, z8, 0);
}

// ---------------------------------------------------------------------------
// Stage 3: out[b][n][o][t] = elu( P @ Wh ),  P_ij = exp(e_ij - m_i) * r_i.
// Block = 8 waves sharing one bt; WhT[bt] streamed through LDS by the TDM in
// double-buffered 64-j chunks (8x less L2 traffic than per-wave VMEM loads).
// ---------------------------------------------------------------------------
__global__ __launch_bounds__(256) void gat_attn_kernel(
    const float* __restrict__ adj, const _Float16* __restrict__ WhT,
    const float* __restrict__ Wh1, const float* __restrict__ Wh2,
    const float* __restrict__ Mrow, const float* __restrict__ Rrow,
    float* __restrict__ out)
{
  __shared__ _Float16 Blds[2][O_ * LROW];      // 2 x 18KB, TDM-padded rows

  const int wid = threadIdx.x >> 5, lane = threadIdx.x & 31;
  const int hf = lane >> 4, l16 = lane & 15;
  const int bt    = blockIdx.x >> 3;           // 8 blocks per bt
  const int igrp  = blockIdx.x & 7;
  const int itile = igrp * 8 + wid;
  const int b = bt >> 3, t = bt & 7;
  const int i = itile * 16 + l16;              // A-matrix row for this lane

  const float wh1i = Wh1[(size_t)bt * N_ + i];
  const float mi   = Mrow[(size_t)bt * N_ + i];
  const float rli  = Rrow[(size_t)bt * N_ + i];
  const float* adjr = adj + (size_t)i * N_;
  const float* wh2r = Wh2 + (size_t)bt * N_;
  const _Float16* whbt = WhT + (size_t)bt * O_ * N_;

  const unsigned int lds0 = (unsigned int)(uintptr_t)&Blds[0][0];
  const unsigned int lds1 = (unsigned int)(uintptr_t)&Blds[1][0];

  v8f acc[8] = {};

  if (wid == 0) tdm_load_chunk(whbt, lds0);    // prefetch chunk 0

  const int NCHUNK = N_ / JCHUNK;              // 16
  for (int c = 0; c < NCHUNK; ++c) {
    const int cur = c & 1;
    if (wid == 0) __builtin_amdgcn_s_wait_tensorcnt(0);
    __syncthreads();                           // chunk c visible to all waves
    if (wid == 0 && (c + 1) < NCHUNK)
      tdm_load_chunk(whbt + (size_t)(c + 1) * JCHUNK, cur ? lds0 : lds1);

    const _Float16* bufc = &Blds[cur][0];

    #pragma unroll
    for (int js = 0; js < JCHUNK; js += 32) {  // 2 sub-chunks of K=32
      const int jb = c * JCHUNK + js;
      const int base0 = jb + 8 * hf;           // lane K groups: base0, base0+16
      float adv[16], wv[16];
      {
        const float4* ap = (const float4*)(adjr + base0);
        const float4* wp = (const float4*)(wh2r + base0);
        float4 q;
        q = ap[0]; adv[0]=q.x;  adv[1]=q.y;  adv[2]=q.z;  adv[3]=q.w;
        q = ap[1]; adv[4]=q.x;  adv[5]=q.y;  adv[6]=q.z;  adv[7]=q.w;
        q = ap[4]; adv[8]=q.x;  adv[9]=q.y;  adv[10]=q.z; adv[11]=q.w;
        q = ap[5]; adv[12]=q.x; adv[13]=q.y; adv[14]=q.z; adv[15]=q.w;
        q = wp[0]; wv[0]=q.x;   wv[1]=q.y;   wv[2]=q.z;   wv[3]=q.w;
        q = wp[1]; wv[4]=q.x;   wv[5]=q.y;   wv[6]=q.z;   wv[7]=q.w;
        q = wp[4]; wv[8]=q.x;   wv[9]=q.y;   wv[10]=q.z;  wv[11]=q.w;
        q = wp[5]; wv[12]=q.x;  wv[13]=q.y;  wv[14]=q.z;  wv[15]=q.w;
      }
      v16h A;
      #pragma unroll
      for (int k = 0; k < 16; ++k) {
        float e = wh1i + wv[k];
        e = fmaxf(e, ALPHA_ * e);              // leaky_relu
        float p = (adv[k] > 0.f) ? __expf(e - mi) * rli : 0.f;
        A[k] = (_Float16)p;
      }

      #pragma unroll
      for (int ot = 0; ot < 8; ++ot) {
        int o = ot * 16 + l16;                 // B column; K = 8*hf + 0..7, 16 + 8*hf + 0..7
        const _Float16* bp = bufc + o * LROW + js + 8 * hf;
        v8h lo = *(const v8h*)bp;              // ds_load_b128
        v8h hi = *(const v8h*)(bp + 16);       // ds_load_b128
        v16h Bv = __builtin_shufflevector(lo, hi,
            0, 1, 2, 3, 4, 5, 6, 7, 8, 9, 10, 11, 12, 13, 14, 15);
        acc[ot] = __builtin_amdgcn_wmma_f32_16x16x32_f16(
            false, A, false, Bv, (short)0, acc[ot], false, false);
      }
    }
    __syncthreads();                           // all reads of buf done
  }

  // Epilogue: elu + scatter to out[b][n][o][t] (t-writes merged in L2).
  const int nb = itile * 16 + 8 * hf;
  #pragma unroll
  for (int ot = 0; ot < 8; ++ot) {
    int o = ot * 16 + l16;
    #pragma unroll
    for (int v = 0; v < 8; ++v) {
      float val = acc[ot][v];
      val = (val > 0.f) ? val : (__expf(val) - 1.f);
      out[(((size_t)b * N_ + nb + v) * O_ + o) * T_ + t] = val;
    }
  }
}

// ---------------------------------------------------------------------------
extern "C" void kernel_launch(void* const* d_in, const int* in_sizes, int n_in,
                              void* d_out, int out_size, void* d_ws, size_t ws_size,
                              hipStream_t stream) {
  (void)in_sizes; (void)n_in; (void)out_size; (void)ws_size;
  const float* x   = (const float*)d_in[0];
  const float* adj = (const float*)d_in[1];
  const float* W   = (const float*)d_in[2];
  const float* a   = (const float*)d_in[3];
  float* out = (float*)d_out;

  char* ws = (char*)d_ws;
  _Float16* WhT = (_Float16*)ws;                               // 16 MB
  size_t off = (size_t)BT_ * O_ * N_ * sizeof(_Float16);
  float* Wh1  = (float*)(ws + off); off += (size_t)BT_ * N_ * sizeof(float);
  float* Wh2  = (float*)(ws + off); off += (size_t)BT_ * N_ * sizeof(float);
  float* Mrow = (float*)(ws + off); off += (size_t)BT_ * N_ * sizeof(float);
  float* Rrow = (float*)(ws + off); off += (size_t)BT_ * N_ * sizeof(float);

  gat_wh_kernel   <<<512,  256, 0, stream>>>(x, W, a, WhT, Wh1, Wh2);
  gat_stats_kernel<<<8192, 256, 0, stream>>>(adj, Wh1, Wh2, Mrow, Rrow);
  gat_attn_kernel <<<512,  256, 0, stream>>>(adj, WhT, Wh1, Wh2, Mrow, Rrow, out);
}
